// AttentionHead_80487687127089
// MI455X (gfx1250) — matrix operand order repaired and checked by reading, not compiled, not run on hardware
//
#include <hip/hip_runtime.h>

// ---------------------------------------------------------------------------
// Causal single-head attention for MI455X (gfx1250, wave32, WMMA).
//   B=4, T=4096, E=1024, H=64, fp32 in/out.
// Pipeline:
//   1) proj_qkv: Q/K/V = x @ W{q,k,v} via v_wmma_f32_16x16x32_bf16,
//      stored as bf16 in workspace; softmax scale E^-0.5 folded into Wq.
//   2) flash_attn: flash-attention forward; 128 query rows per workgroup,
//      4 waves x 2 strips of 16 rows; each LDS B-fragment feeds 2 WMMAs.
//      fp32 online softmax with shfl_xor reductions in 16-lane halves.
// ---------------------------------------------------------------------------

#define T_SEQ 4096
#define E_DIM 1024
#define H_DIM 64
#define NBATCH 4

typedef __attribute__((ext_vector_type(16))) __bf16 v16bf;
typedef __attribute__((ext_vector_type(8)))  float  v8f;

union ABFrag { v16bf v; uint4 q[2]; };

// Native f32 -> bf16 (RNE) — let clang use gfx1250 hardware converts.
__device__ __forceinline__ unsigned short f2bf(float f) {
    union { __bf16 b; unsigned short u; } c;
    c.b = (__bf16)f;
    return c.u;
}

__device__ __forceinline__ v8f wmma_bf16(const ABFrag& a, const ABFrag& b, v8f c) {
    // (neg_a, A, neg_b, B, c_mod, C, reuse_a, reuse_b)
    return __builtin_amdgcn_wmma_f32_16x16x32_bf16(false, a.v, false, b.v,
                                                   (short)0, c, false, false);
}

__device__ __forceinline__ v8f v8f_zero() {
    v8f z = {0.f, 0.f, 0.f, 0.f, 0.f, 0.f, 0.f, 0.f};
    return z;
}

// ---------------------------------------------------------------------------
// Projection kernel: 128 rows of x per workgroup, 8 waves (256 threads).
// Each wave owns a 16-row strip; computes K/Q/V (3 outputs x 4 n-tiles).
// K-loop: E=1024 in chunks of 32 (one 16x16x32 WMMA chain step).
// ---------------------------------------------------------------------------
__global__ __launch_bounds__(256) void proj_qkv(
    const float* __restrict__ x,
    const float* __restrict__ Wk,
    const float* __restrict__ Wq,
    const float* __restrict__ Wv,
    unsigned short* __restrict__ Ko,   // bf16 [B*T][64]
    unsigned short* __restrict__ Qo,   // bf16 (pre-scaled by E^-0.5)
    unsigned short* __restrict__ Vo)   // bf16
{
    __shared__ unsigned short xs[128][40];       // x chunk as bf16, [m][k], pad 40
    __shared__ unsigned short wsm[3][64][40];    // W^T chunks, [n][k], pad 40

    const int tid  = threadIdx.x;
    const int wv   = tid >> 5;
    const int lane = tid & 31;
    const int mrow = lane & 15;
    const int half = lane >> 4;
    const long row0 = (long)blockIdx.x * 128;

    v8f acc[3][4];
#pragma unroll
    for (int w = 0; w < 3; ++w)
#pragma unroll
        for (int nt = 0; nt < 4; ++nt) acc[w][nt] = v8f_zero();

    for (int kc = 0; kc < E_DIM; kc += 32) {
        // ---- stage x chunk (128x32 f32 -> bf16) ----
        {
            const int r  = tid >> 1;
            const int cb = (tid & 1) * 16;
            const float* src = x + (row0 + r) * E_DIM + kc + cb;
#pragma unroll
            for (int j = 0; j < 16; j += 4) {
                float4 f = *(const float4*)(src + j);
                xs[r][cb + j + 0] = f2bf(f.x);
                xs[r][cb + j + 1] = f2bf(f.y);
                xs[r][cb + j + 2] = f2bf(f.z);
                xs[r][cb + j + 3] = f2bf(f.w);
            }
        }
        // ---- stage W^T chunks (32x64 f32 -> bf16 [n][k]), scale folded into Wq ----
        {
            const float* wp[3] = {Wk, Wq, Wv};
#pragma unroll
            for (int w = 0; w < 3; ++w) {
#pragma unroll
                for (int j = 0; j < 8; ++j) {
                    int flat = tid * 8 + j;          // 0..2047
                    int e = flat >> 6;               // k within chunk
                    int h = flat & 63;               // n
                    float val = wp[w][(long)(kc + e) * H_DIM + h];
                    if (w == 1) val *= 0.03125f;     // E^-0.5 = 1/32
                    wsm[w][h][e] = f2bf(val);
                }
            }
        }
        __syncthreads();

        // ---- A fragment: 16x32 strip of x for this wave ----
        ABFrag a;
        const unsigned short* abase = &xs[wv * 16 + mrow][0];
        a.q[0] = *(const uint4*)(abase + 8 * half);
        a.q[1] = *(const uint4*)(abase + 16 + 8 * half);

        // ---- 12 WMMAs per chunk per wave ----
#pragma unroll
        for (int w = 0; w < 3; ++w) {
#pragma unroll
            for (int nt = 0; nt < 4; ++nt) {
                ABFrag b;
                const unsigned short* bbase = &wsm[w][nt * 16 + mrow][0];
                b.q[0] = *(const uint4*)(bbase + 8 * half);
                b.q[1] = *(const uint4*)(bbase + 16 + 8 * half);
                acc[w][nt] = wmma_bf16(a, b, acc[w][nt]);
            }
        }
        __syncthreads();
    }

    // ---- store bf16 results: C layout row = i + 8*half, col = nt*16 + mrow ----
    unsigned short* outp[3] = {Ko, Qo, Vo};
#pragma unroll
    for (int w = 0; w < 3; ++w) {
#pragma unroll
        for (int nt = 0; nt < 4; ++nt) {
#pragma unroll
            for (int i = 0; i < 8; ++i) {
                long row = row0 + wv * 16 + i + 8 * half;
                outp[w][row * H_DIM + nt * 16 + mrow] = f2bf(acc[w][nt][i]);
            }
        }
    }
}

// ---------------------------------------------------------------------------
// Flash attention: grid (T/128, B), 4 waves (128 threads) per workgroup.
// Each wave owns TWO 16-row strips (rows wv*32 and wv*32+16) of the 128-row
// query block, so each K/V B-fragment load feeds two WMMAs.
// KV blocks of 64; fully-masked (strip, block) pairs are skipped.
// ---------------------------------------------------------------------------
__global__ __launch_bounds__(128) void flash_attn(
    const unsigned short* __restrict__ Q,   // bf16 [B*T][64], pre-scaled
    const unsigned short* __restrict__ Kp,  // bf16 [B*T][64]
    const unsigned short* __restrict__ Vp,  // bf16 [B*T][64]
    float* __restrict__ Out)                // f32 [B*T][64]
{
    __shared__ unsigned short Ks[64][72];        // K tile [t][h] (== [n][k] for QK^T)
    __shared__ unsigned short Vt[64][72];        // V tile transposed [h][t]
    __shared__ unsigned short Ps[4][2][16][64];  // per-wave/strip P tiles

    const int tid  = threadIdx.x;
    const int wv   = tid >> 5;
    const int lane = tid & 31;
    const int mrow = lane & 15;
    const int half = lane >> 4;
    const int b    = blockIdx.y;
    const int bx   = blockIdx.x;
    const int qb   = bx * 128;
    const int jmax = 2 * bx + 1;                 // last (diagonal) KV block
    const long boff = (long)b * T_SEQ * H_DIM;

    // ---- Q fragments: 2 strips x (K=0..63 -> 2 frags) ----
    ABFrag qa[2][2];
#pragma unroll
    for (int st = 0; st < 2; ++st) {
        const int sb = wv * 32 + st * 16;
        const unsigned short* qbase = Q + boff + (long)(qb + sb + mrow) * H_DIM;
        qa[st][0].q[0] = *(const uint4*)(qbase + 8 * half);
        qa[st][0].q[1] = *(const uint4*)(qbase + 16 + 8 * half);
        qa[st][1].q[0] = *(const uint4*)(qbase + 32 + 8 * half);
        qa[st][1].q[1] = *(const uint4*)(qbase + 48 + 8 * half);
    }

    v8f acc[2][4];
    float mi[2][8], li[2][8];
#pragma unroll
    for (int st = 0; st < 2; ++st) {
#pragma unroll
        for (int nt = 0; nt < 4; ++nt) acc[st][nt] = v8f_zero();
#pragma unroll
        for (int i = 0; i < 8; ++i) { mi[st][i] = -1e30f; li[st][i] = 0.f; }
    }

    for (int j = 0; j <= jmax; ++j) {
        const long kvoff = boff + (long)j * 64 * H_DIM;
        // ---- stage K tile (row-major) and V tile (transposed) ----
        {
            const int r  = tid >> 1;
            const int cb = (tid & 1) * 32;
            const uint4* ksrc = (const uint4*)(Kp + kvoff + (long)r * H_DIM + cb);
            uint4* kdst = (uint4*)&Ks[r][cb];
            kdst[0] = ksrc[0]; kdst[1] = ksrc[1]; kdst[2] = ksrc[2]; kdst[3] = ksrc[3];

            const uint4* vsrc = (const uint4*)(Vp + kvoff + (long)r * H_DIM + cb);
            union { uint4 q[4]; unsigned short s[32]; } vtmp;
            vtmp.q[0] = vsrc[0]; vtmp.q[1] = vsrc[1];
            vtmp.q[2] = vsrc[2]; vtmp.q[3] = vsrc[3];
#pragma unroll
            for (int c = 0; c < 32; ++c) Vt[cb + c][r] = vtmp.s[c];

            // prefetch next KV tile into caches (global_prefetch_b8)
            if (j < jmax) {
                __builtin_prefetch(Kp + kvoff + 64 * H_DIM + (long)r * H_DIM + cb, 0, 1);
                __builtin_prefetch(Vp + kvoff + 64 * H_DIM + (long)r * H_DIM + cb, 0, 1);
            }
        }
        __syncthreads();

        // Strip visibility: strip rows [qb+sb, qb+sb+15]; block cols start j*64.
        bool vis[2];
#pragma unroll
        for (int st = 0; st < 2; ++st)
            vis[st] = (j * 64 <= qb + wv * 32 + st * 16 + 15);
        const bool need_mask0 = (j * 64 + 63 > qb + wv * 32);
        const bool need_mask1 = (j * 64 + 63 > qb + wv * 32 + 16);

        // ---- S = Q @ K^T : each B fragment feeds both strips ----
        v8f s[2][4];
#pragma unroll
        for (int nt = 0; nt < 4; ++nt) {
            ABFrag b0, b1;
            const unsigned short* kb = &Ks[nt * 16 + mrow][0];
            b0.q[0] = *(const uint4*)(kb + 8 * half);
            b0.q[1] = *(const uint4*)(kb + 16 + 8 * half);
            b1.q[0] = *(const uint4*)(kb + 32 + 8 * half);
            b1.q[1] = *(const uint4*)(kb + 48 + 8 * half);
#pragma unroll
            for (int st = 0; st < 2; ++st) {
                if (vis[st]) {
                    s[st][nt] = wmma_bf16(qa[st][0], b0, v8f_zero());
                    s[st][nt] = wmma_bf16(qa[st][1], b1, s[st][nt]);
                }
            }
        }

        // ---- causal mask + online softmax + P store, per strip ----
#pragma unroll
        for (int st = 0; st < 2; ++st) {
            if (!vis[st]) continue;
            const int sb = wv * 32 + st * 16;
            const bool need_mask = (st == 0) ? need_mask0 : need_mask1;
            if (need_mask) {
#pragma unroll
                for (int nt = 0; nt < 4; ++nt)
#pragma unroll
                    for (int i = 0; i < 8; ++i) {
                        int row = qb + sb + i + 8 * half;
                        int col = j * 64 + nt * 16 + mrow;
                        if (col > row) s[st][nt][i] = -1e30f;
                    }
            }
#pragma unroll
            for (int i = 0; i < 8; ++i) {
                float rm = fmaxf(fmaxf(s[st][0][i], s[st][1][i]),
                                 fmaxf(s[st][2][i], s[st][3][i]));
                rm = fmaxf(rm, __shfl_xor(rm, 1, 32));
                rm = fmaxf(rm, __shfl_xor(rm, 2, 32));
                rm = fmaxf(rm, __shfl_xor(rm, 4, 32));
                rm = fmaxf(rm, __shfl_xor(rm, 8, 32));
                float mnew  = fmaxf(mi[st][i], rm);
                float alpha = __expf(mi[st][i] - mnew);
                float psum  = 0.f;
                const int m = i + 8 * half;
#pragma unroll
                for (int nt = 0; nt < 4; ++nt) {
                    float p = __expf(s[st][nt][i] - mnew);
                    psum += p;
                    Ps[wv][st][m][nt * 16 + mrow] = f2bf(p);
                }
                psum += __shfl_xor(psum, 1, 32);
                psum += __shfl_xor(psum, 2, 32);
                psum += __shfl_xor(psum, 4, 32);
                psum += __shfl_xor(psum, 8, 32);
                li[st][i] = li[st][i] * alpha + psum;
                mi[st][i] = mnew;
#pragma unroll
                for (int nt = 0; nt < 4; ++nt) acc[st][nt][i] *= alpha;
            }
        }

        // ---- O += P @ V : each V B-fragment feeds both strips ----
        ABFrag pa[2][2];
#pragma unroll
        for (int st = 0; st < 2; ++st) {
            if (!vis[st]) continue;
            const unsigned short* pb = &Ps[wv][st][mrow][0];
            pa[st][0].q[0] = *(const uint4*)(pb + 8 * half);
            pa[st][0].q[1] = *(const uint4*)(pb + 16 + 8 * half);
            pa[st][1].q[0] = *(const uint4*)(pb + 32 + 8 * half);
            pa[st][1].q[1] = *(const uint4*)(pb + 48 + 8 * half);
        }
#pragma unroll
        for (int nt = 0; nt < 4; ++nt) {
            ABFrag b0, b1;
            const unsigned short* vb = &Vt[nt * 16 + mrow][0];
            b0.q[0] = *(const uint4*)(vb + 8 * half);
            b0.q[1] = *(const uint4*)(vb + 16 + 8 * half);
            b1.q[0] = *(const uint4*)(vb + 32 + 8 * half);
            b1.q[1] = *(const uint4*)(vb + 48 + 8 * half);
#pragma unroll
            for (int st = 0; st < 2; ++st) {
                if (vis[st]) {
                    acc[st][nt] = wmma_bf16(pa[st][0], b0, acc[st][nt]);
                    acc[st][nt] = wmma_bf16(pa[st][1], b1, acc[st][nt]);
                }
            }
        }
        __syncthreads();
    }

    // ---- epilogue: normalize and store f32 ----
#pragma unroll
    for (int st = 0; st < 2; ++st) {
        const int sb = wv * 32 + st * 16;
#pragma unroll
        for (int nt = 0; nt < 4; ++nt) {
#pragma unroll
            for (int i = 0; i < 8; ++i) {
                long row = qb + sb + i + 8 * half;
                Out[boff + row * H_DIM + nt * 16 + mrow] = acc[st][nt][i] / li[st][i];
            }
        }
    }
}

// ---------------------------------------------------------------------------
extern "C" void kernel_launch(void* const* d_in, const int* in_sizes, int n_in,
                              void* d_out, int out_size, void* d_ws, size_t ws_size,
                              hipStream_t stream) {
    const float* x  = (const float*)d_in[0];
    const float* Wk = (const float*)d_in[1];
    const float* Wq = (const float*)d_in[2];
    const float* Wv = (const float*)d_in[3];
    float* out = (float*)d_out;

    const size_t bt = (size_t)NBATCH * T_SEQ;          // 16384 rows
    unsigned short* Kw = (unsigned short*)d_ws;        // bf16 [B*T][64]
    unsigned short* Qw = Kw + bt * H_DIM;
    unsigned short* Vw = Qw + bt * H_DIM;

    // 1) projections: 16384 rows / 128 per block = 128 blocks, 8 waves each
    proj_qkv<<<dim3((unsigned)(bt / 128)), 256, 0, stream>>>(x, Wk, Wq, Wv, Kw, Qw, Vw);

    // 2) flash attention: (T/128, B) blocks, 4 waves each
    flash_attn<<<dim3(T_SEQ / 128, NBATCH), 128, 0, stream>>>(Qw, Kw, Vw, out);
}
